// NeuralODE_36807869727439
// MI455X (gfx1250) — compile-verified, matrix operand order checked by reading, not run on hardware
//
#include <hip/hip_runtime.h>
#include <math.h>
#include <stdint.h>

// ---------------------------------------------------------------------------
// Neural ODE on MI455X (gfx1250):
//   20 Euler steps of  z += h * ( tanh(z@W1 + b1 + t*u) @ W2 + b2 )
// bf16 WMMA (v_wmma_f32_16x16x32_bf16) GEMMs with f32 accumulation.
// Weights converted fp32->bf16 and pre-transposed once per launch.
// GEMM mainloop double-buffers LDS; bf16 tiles are moved with the CDNA5
// async copy engine (global_load_async_to_lds_b128, ASYNCcnt), fp32 z tiles
// are converted in-flight through VGPRs.
// ---------------------------------------------------------------------------

typedef __bf16 bf16_t;
typedef __attribute__((ext_vector_type(16))) __bf16 v16bf;
typedef __attribute__((ext_vector_type(8)))  __bf16 v8bf;
typedef __attribute__((ext_vector_type(8)))  float  v8f;
typedef __attribute__((ext_vector_type(4)))  float  v4f;

#define BATCH 256
#define DDIM  2048
#define HDIM  4096
#define NSUB  2        // ceil(0.1 / 0.05) Euler substeps per interval
#define NIVAL 10       // T-1 intervals

// GEMM tiling: 64x128 C tile per block, K-chunk 32, 256 threads = 8 waves(2x4)
#define BM 64
#define BN 128
#define BK 32
#define LDSS 40        // padded LDS row stride in bf16 elements (80B)

// ---- CDNA5 async copy: 16B global -> LDS, tracked by ASYNCcnt --------------
__device__ __forceinline__ void async_copy_b128(void* lds_ptr, const void* gptr) {
  // VDST = LDS byte address (low 32 bits of generic pointer, per flat->LDS
  // truncation), VADDR = 64-bit global address, SADDR = off (GV mode).
  asm volatile("global_load_async_to_lds_b128 %0, %1, off"
               :: "v"((uint32_t)(uintptr_t)lds_ptr), "v"(gptr)
               : "memory");
}
__device__ __forceinline__ void wait_async0() {
  asm volatile("s_wait_asynccnt 0x0" ::: "memory");
}

// ---- fp32 -> bf16 convert + transpose (weights, once per launch) -----------
// src: K x N fp32 (row-major). dst: N x K bf16 (row-major), i.e. B^T.
#define TT 32
__global__ __launch_bounds__(256) void cvt_transpose_bf16(
    const float* __restrict__ src, bf16_t* __restrict__ dst, int K, int N) {
  __shared__ bf16_t tile[TT][TT + 2];
  const int n0 = blockIdx.x * TT;
  const int k0 = blockIdx.y * TT;
  const int tx = threadIdx.x & 31;
  const int ty = threadIdx.x >> 5;  // 0..7
#pragma unroll
  for (int r = 0; r < TT; r += 8) {
    float v = src[(size_t)(k0 + ty + r) * N + n0 + tx];
    tile[ty + r][tx] = (bf16_t)v;
  }
  __syncthreads();
#pragma unroll
  for (int r = 0; r < TT; r += 8) {
    dst[(size_t)(n0 + ty + r) * K + k0 + tx] = tile[tx][ty + r];
  }
}

// ---- shared WMMA mainloop (double-buffered LDS, async B/A copies) -----------
// A: MxK row-major (fp32 or bf16). Bt: NxK row-major bf16 (pre-transposed).
// Each wave accumulates a 32x32 C patch = 2x2 wmma 16x16 tiles.
template <bool A_F32>
__device__ __forceinline__ void gemm_tile(const void* __restrict__ Ag, int lda,
                                          const bf16_t* __restrict__ Bt, int ldbt,
                                          int K, int m_base, int n_base,
                                          bf16_t* sA, bf16_t* sB, v8f acc[2][2]) {
  const int tid  = threadIdx.x;
  const int lane = tid & 31;
  const int wave = tid >> 5;
  const int wm = wave >> 2;          // 0..1  (M sub-block of 32)
  const int wn = wave & 3;           // 0..3  (N sub-block of 32)
  const int fm = lane & 15;          // fragment row within 16
  const int kb = (lane >> 4) << 3;   // lane-half K base: 0 or 8

  // global->LDS copy mapping (all 16B-aligned)
  const int ar  = tid >> 2;          // 0..63   A row
  const int ac  = (tid & 3) * 8;     // A col (8 bf16 per thread)
  const int bn  = tid >> 1;          // 0..127  B^T row (= n)
  const int bkc = (tid & 1) * 16;    // B^T col (16 bf16 per thread)

  const float*  Af = (const float*)Ag;
  const bf16_t* Ab = (const bf16_t*)Ag;

  v8bf ra;  // fp32-A staging only

  // ---- issue async copies of tile kk into LDS buffer `buf`
  auto issue_async = [&](int kk, int buf) {
    bf16_t* b = sB + buf * (BN * LDSS);
    const bf16_t* pb = Bt + (size_t)(n_base + bn) * ldbt + kk + bkc;
    async_copy_b128(b + bn * LDSS + bkc, pb);
    async_copy_b128(b + bn * LDSS + bkc + 8, pb + 8);
    if (!A_F32) {
      bf16_t* a = sA + buf * (BM * LDSS);
      async_copy_b128(a + ar * LDSS + ac,
                      Ab + (size_t)(m_base + ar) * lda + kk + ac);
    }
  };
  // ---- fp32 A tile: load + convert through VGPRs
  auto loadA_f32 = [&](int kk) {
    const float* pa = Af + (size_t)(m_base + ar) * lda + kk + ac;
    v4f a0 = *(const v4f*)pa;
    v4f a1 = *(const v4f*)(pa + 4);
    ra[0] = (bf16_t)a0[0]; ra[1] = (bf16_t)a0[1]; ra[2] = (bf16_t)a0[2]; ra[3] = (bf16_t)a0[3];
    ra[4] = (bf16_t)a1[0]; ra[5] = (bf16_t)a1[1]; ra[6] = (bf16_t)a1[2]; ra[7] = (bf16_t)a1[3];
  };
  auto stageA = [&](int buf) {
    *(v8bf*)(sA + buf * (BM * LDSS) + ar * LDSS + ac) = ra;
  };

  issue_async(0, 0);
  if (A_F32) { loadA_f32(0); stageA(0); }
  wait_async0();
  __syncthreads();

  int cur = 0;
  for (int kk = 0; kk < K; kk += BK) {
    const bool has_next = (kk + BK) < K;
    if (has_next) {
      issue_async(kk + BK, cur ^ 1);
      if (A_F32) {
        loadA_f32(kk + BK);
        if (kk + 2 * BK < K)
          __builtin_prefetch(Af + (size_t)(m_base + ar) * lda + kk + 2 * BK + ac, 0, 1);
      }
    }

    // ---- fragments from buffer `cur` (ISA 16-bit 16x32 layout:
    //      elems 0..7 = K kb..kb+7, elems 8..15 = K kb+16..kb+23)
    const bf16_t* a = sA + cur * (BM * LDSS);
    const bf16_t* b = sB + cur * (BN * LDSS);
    v16bf aF[2], bF[2];
#pragma unroll
    for (int i = 0; i < 2; ++i) {
      const bf16_t* p = a + (wm * 32 + i * 16 + fm) * LDSS + kb;
      v8bf lo = *(const v8bf*)p;
      v8bf hi = *(const v8bf*)(p + 16);
      aF[i] = __builtin_shufflevector(lo, hi, 0,1,2,3,4,5,6,7,8,9,10,11,12,13,14,15);
    }
#pragma unroll
    for (int j = 0; j < 2; ++j) {
      const bf16_t* p = b + (wn * 32 + j * 16 + fm) * LDSS + kb;
      v8bf lo = *(const v8bf*)p;
      v8bf hi = *(const v8bf*)(p + 16);
      bF[j] = __builtin_shufflevector(lo, hi, 0,1,2,3,4,5,6,7,8,9,10,11,12,13,14,15);
    }
#pragma unroll
    for (int i = 0; i < 2; ++i)
#pragma unroll
      for (int j = 0; j < 2; ++j)
        acc[i][j] = __builtin_amdgcn_wmma_f32_16x16x32_bf16(
            false, aF[i], false, bF[j], (short)0, acc[i][j], false, false);

    if (A_F32 && has_next) stageA(cur ^ 1);
    wait_async0();          // this wave's async copies into cur^1 are in LDS
    __syncthreads();        // all waves' copies visible; reads of `cur` done
    cur ^= 1;
  }
}

// ---- GEMM1: act = tanh(z @ W1 + b1 + ti*u), bf16 output ---------------------
__global__ __launch_bounds__(256) void gemm1_tanh(
    const float* __restrict__ z, const bf16_t* __restrict__ W1t,
    const float* __restrict__ b1, const float* __restrict__ u,
    const float* __restrict__ tgrid, int iv, int sub,
    bf16_t* __restrict__ act) {
  __shared__ bf16_t sA[2 * BM * LDSS];
  __shared__ bf16_t sB[2 * BN * LDSS];
  const int m_base = blockIdx.y * BM;
  const int n_base = blockIdx.x * BN;
  v8f acc[2][2];
#pragma unroll
  for (int i = 0; i < 2; ++i)
#pragma unroll
    for (int j = 0; j < 2; ++j)
#pragma unroll
      for (int e = 0; e < 8; ++e) acc[i][j][e] = 0.0f;

  gemm_tile<true>(z, DDIM, W1t, DDIM, DDIM, m_base, n_base, sA, sB, acc);

  const float t0 = tgrid[iv];
  const float t1 = tgrid[iv + 1];
  const float hh = (t1 - t0) * (1.0f / NSUB);
  const float ti = t0 + hh * (float)(sub + 1);  // reference updates t BEFORE f()

  const int lane = threadIdx.x & 31;
  const int wave = threadIdx.x >> 5;
  const int wm = wave >> 2, wn = wave & 3;
  const int nl = lane & 15;
  const int ro = (lane >> 4) * 8;  // C layout: VGPR r -> row r (lanes<16) / r+8
#pragma unroll
  for (int j = 0; j < 2; ++j) {
    const int gn = n_base + wn * 32 + j * 16 + nl;
    const float bias = b1[gn] + ti * u[gn];
#pragma unroll
    for (int i = 0; i < 2; ++i) {
      const int gm0 = m_base + wm * 32 + i * 16 + ro;
#pragma unroll
      for (int r = 0; r < 8; ++r) {
        float v = acc[i][j][r] + bias;
        act[(size_t)(gm0 + r) * HDIM + gn] = (bf16_t)tanhf(v);
      }
    }
  }
}

// ---- GEMM2: zdst = zsrc + h * (act @ W2 + b2), fp32 output ------------------
__global__ __launch_bounds__(256) void gemm2_update(
    const bf16_t* __restrict__ act, const bf16_t* __restrict__ W2t,
    const float* __restrict__ b2, const float* __restrict__ zsrc,
    const float* __restrict__ tgrid, int iv,
    float* __restrict__ zdst) {
  __shared__ bf16_t sA[2 * BM * LDSS];
  __shared__ bf16_t sB[2 * BN * LDSS];
  const int m_base = blockIdx.y * BM;
  const int n_base = blockIdx.x * BN;
  v8f acc[2][2];
#pragma unroll
  for (int i = 0; i < 2; ++i)
#pragma unroll
    for (int j = 0; j < 2; ++j)
#pragma unroll
      for (int e = 0; e < 8; ++e) acc[i][j][e] = 0.0f;

  gemm_tile<false>(act, HDIM, W2t, HDIM, HDIM, m_base, n_base, sA, sB, acc);

  const float t0 = tgrid[iv];
  const float t1 = tgrid[iv + 1];
  const float hh = (t1 - t0) * (1.0f / NSUB);

  const int lane = threadIdx.x & 31;
  const int wave = threadIdx.x >> 5;
  const int wm = wave >> 2, wn = wave & 3;
  const int nl = lane & 15;
  const int ro = (lane >> 4) * 8;
#pragma unroll
  for (int j = 0; j < 2; ++j) {
    const int gn = n_base + wn * 32 + j * 16 + nl;
    const float bias = b2[gn];
#pragma unroll
    for (int i = 0; i < 2; ++i) {
      const int gm0 = m_base + wm * 32 + i * 16 + ro;
#pragma unroll
      for (int r = 0; r < 8; ++r) {
        const size_t idx = (size_t)(gm0 + r) * DDIM + gn;
        zdst[idx] = zsrc[idx] + hh * (acc[i][j][r] + bias);
      }
    }
  }
}

// ---------------------------------------------------------------------------
extern "C" void kernel_launch(void* const* d_in, const int* in_sizes, int n_in,
                              void* d_out, int out_size, void* d_ws, size_t ws_size,
                              hipStream_t stream) {
  const float* z0    = (const float*)d_in[0];
  const float* tgrid = (const float*)d_in[1];
  const float* W1    = (const float*)d_in[2];
  const float* b1    = (const float*)d_in[3];
  const float* u     = (const float*)d_in[4];
  const float* W2    = (const float*)d_in[5];
  const float* b2    = (const float*)d_in[6];
  float* out = (float*)d_out;

  char* ws = (char*)d_ws;
  const size_t wbytes = (size_t)DDIM * HDIM * sizeof(bf16_t);   // 16 MB
  const size_t abytes = (size_t)BATCH * HDIM * sizeof(bf16_t);  // 2 MB
  const size_t zbytes = (size_t)BATCH * DDIM * sizeof(float);   // 2 MB
  bf16_t* W1t  = (bf16_t*)(ws);                     // H x D  (= W1^T)
  bf16_t* W2t  = (bf16_t*)(ws + wbytes);            // D x H  (= W2^T)
  bf16_t* actb = (bf16_t*)(ws + 2 * wbytes);
  float*  zA   = (float*)(ws + 2 * wbytes + abytes);
  float*  zB   = (float*)(ws + 2 * wbytes + abytes + zbytes);

  // one-time weight downconversion + transpose (deterministic every launch)
  cvt_transpose_bf16<<<dim3(HDIM / TT, DDIM / TT), 256, 0, stream>>>(W1, W1t, DDIM, HDIM);
  cvt_transpose_bf16<<<dim3(DDIM / TT, HDIM / TT), 256, 0, stream>>>(W2, W2t, HDIM, DDIM);

  const dim3 grid1(HDIM / BN, BATCH / BM);  // 32 x 4
  const dim3 grid2(DDIM / BN, BATCH / BM);  // 16 x 4
  const int nsteps = NIVAL * NSUB;          // 20

  for (int k = 0; k < nsteps; ++k) {
    const int iv  = k / NSUB;
    const int sub = k % NSUB;
    const float* zsrc = (k == 0) ? z0 : (((k - 1) & 1) ? zB : zA);
    float* zdst = (k == nsteps - 1) ? out : ((k & 1) ? zB : zA);
    gemm1_tanh<<<grid1, 256, 0, stream>>>(zsrc, W1t, b1, u, tgrid, iv, sub, actb);
    gemm2_update<<<grid2, 256, 0, stream>>>(actb, W2t, b2, zsrc, tgrid, iv, zdst);
  }
}
